// RWKVAttention_v2_84241488544515
// MI455X (gfx1250) — matrix operand order repaired and checked by reading, not compiled, not run on hardware
//
#include <hip/hip_runtime.h>
#include <cstdint>

#define BDIM   1024
#define TLEN   2048
#define NBATCH 8
#define MROWS  (NBATCH * TLEN)   // 16384

typedef __attribute__((ext_vector_type(16))) __bf16 v16bf;
typedef __attribute__((ext_vector_type(8)))  float  v8f;

union Frag { __bf16 h[16]; unsigned d[8]; v16bf v; };

// ---------------- prep: fp32 -> bf16 (x) ----------------
__global__ __launch_bounds__(256) void f32_to_bf16_kernel(
    const float* __restrict__ in, __bf16* __restrict__ out)
{
    size_t i = ((size_t)blockIdx.x * 256 + threadIdx.x) * 8;
    float4 a = *(const float4*)(in + i);
    float4 b = *(const float4*)(in + i + 4);
    union { __bf16 h[8]; uint4 u; } p;
    p.h[0] = (__bf16)a.x; p.h[1] = (__bf16)a.y; p.h[2] = (__bf16)a.z; p.h[3] = (__bf16)a.w;
    p.h[4] = (__bf16)b.x; p.h[5] = (__bf16)b.y; p.h[6] = (__bf16)b.z; p.h[7] = (__bf16)b.w;
    *(uint4*)(out + i) = p.u;
}

// ---------------- prep: transpose W [K][N] fp32 -> Wt [N][K] bf16 ----------------
__global__ __launch_bounds__(256) void transpose_f32_bf16(
    const float* __restrict__ W, __bf16* __restrict__ Wt)
{
    __shared__ float tile[32][33];
    int kb = blockIdx.x * 32;
    int nb = blockIdx.y * 32;
    int tx = threadIdx.x & 31;
    int ty = threadIdx.x >> 5;   // 0..7
    #pragma unroll
    for (int i = 0; i < 4; ++i)
        tile[ty + i * 8][tx] = W[(size_t)(kb + ty + i * 8) * BDIM + nb + tx];
    __syncthreads();
    #pragma unroll
    for (int i = 0; i < 4; ++i)
        Wt[(size_t)(nb + ty + i * 8) * BDIM + kb + tx] = (__bf16)tile[tx][ty + i * 8];
}

// ---------------- GEMM: out[M,1024] = A[M,K]bf16 @ Bt[N,K]bf16^T + bias ----------------
// Block tile 128x128, BK=32, 8 waves (2 M-groups x 4 N-groups), wave tile 64x32.
// Double-buffered LDS fed by global_load_async_to_lds_b128 (ASYNCcnt pipelined).
__global__ __launch_bounds__(256) void gemm_bf16_tn(
    const __bf16* __restrict__ A,   // [M][1024] bf16 row-major
    const __bf16* __restrict__ Bt,  // [1024][1024] bf16, row n = weight column n
    const float* __restrict__ bias,
    float* __restrict__ out)
{
    __shared__ __bf16 As[2][128 * 32];  // [row][k]
    __shared__ __bf16 Bs[2][128 * 32];  // [n][k]

    const int tid  = threadIdx.x;
    const int lane = tid & 31;
    const int wid  = tid >> 5;
    const int wm   = wid & 1;          // 64-row group
    const int wn   = wid >> 1;         // 32-col group
    const int row0 = blockIdx.y * 128;
    const int col0 = blockIdx.x * 128;

    // copy mapping: tile = 128 rows x 64B; 512 x 16B chunks; 2 chunks per thread
    const int r0 = tid >> 2,          q0 = tid & 3;
    const int r1 = (tid + 256) >> 2,  q1 = (tid + 256) & 3;

    const uint64_t baseA = (uint64_t)(uintptr_t)A;
    const uint64_t baseB = (uint64_t)(uintptr_t)Bt;

    auto issue_tile = [&](int buf, int kk) {
        unsigned lA0 = (unsigned)(uintptr_t)&As[buf][r0 * 32 + q0 * 8];
        unsigned lA1 = (unsigned)(uintptr_t)&As[buf][r1 * 32 + q1 * 8];
        unsigned lB0 = (unsigned)(uintptr_t)&Bs[buf][r0 * 32 + q0 * 8];
        unsigned lB1 = (unsigned)(uintptr_t)&Bs[buf][r1 * 32 + q1 * 8];
        unsigned oA0 = (unsigned)(((row0 + r0) * BDIM + kk + q0 * 8) * 2);
        unsigned oA1 = (unsigned)(((row0 + r1) * BDIM + kk + q1 * 8) * 2);
        unsigned oB0 = (unsigned)(((col0 + r0) * BDIM + kk + q0 * 8) * 2);
        unsigned oB1 = (unsigned)(((col0 + r1) * BDIM + kk + q1 * 8) * 2);
        asm volatile("global_load_async_to_lds_b128 %0, %1, %2"
                     :: "v"(lA0), "v"(oA0), "s"(baseA) : "memory");
        asm volatile("global_load_async_to_lds_b128 %0, %1, %2"
                     :: "v"(lA1), "v"(oA1), "s"(baseA) : "memory");
        asm volatile("global_load_async_to_lds_b128 %0, %1, %2"
                     :: "v"(lB0), "v"(oB0), "s"(baseB) : "memory");
        asm volatile("global_load_async_to_lds_b128 %0, %1, %2"
                     :: "v"(lB1), "v"(oB1), "s"(baseB) : "memory");
    };

    v8f acc[4][2];
    #pragma unroll
    for (int i = 0; i < 4; ++i)
        #pragma unroll
        for (int j = 0; j < 2; ++j)
            acc[i][j] = (v8f){0.f,0.f,0.f,0.f,0.f,0.f,0.f,0.f};

    issue_tile(0, 0);

    const int m    = lane & 15;
    const int half = lane >> 4;

    for (int kt = 0; kt < BDIM / 32; ++kt) {
        const int cur = kt & 1;
        if (kt + 1 < BDIM / 32) {
            issue_tile(cur ^ 1, (kt + 1) * 32);
            // 4 newer copies may stay in flight; current tile's 4 (older, in-order) done
            asm volatile("s_wait_asynccnt 0x4" ::: "memory");
        } else {
            asm volatile("s_wait_asynccnt 0x0" ::: "memory");
        }
        __syncthreads();

        Frag fa[4], fb[2];
        // A operand 16x32: v<4 -> K = half*8 + 2v ; v>=4 -> K = 16 + half*8 + 2(v-4)
        #pragma unroll
        for (int am = 0; am < 4; ++am) {
            const __bf16* ar = &As[cur][(wm * 64 + am * 16 + m) * 32];
            #pragma unroll
            for (int v2 = 0; v2 < 8; ++v2) {
                int kidx = (v2 < 4) ? (half * 8 + 2 * v2) : (16 + half * 8 + 2 * (v2 - 4));
                fa[am].d[v2] = *(const unsigned*)(ar + kidx);
            }
        }
        // B operand 32x16: lane=col | v<4 -> K = half*8 + 2v ; v>=4 -> K = 16 + half*8 + 2(v-4)
        #pragma unroll
        for (int bn = 0; bn < 2; ++bn) {
            const __bf16* bc = &Bs[cur][(wn * 32 + bn * 16 + m) * 32];
            #pragma unroll
            for (int v2 = 0; v2 < 8; ++v2) {
                int kidx = (v2 < 4) ? (half * 8 + 2 * v2) : (16 + half * 8 + 2 * (v2 - 4));
                fb[bn].d[v2] = *(const unsigned*)(bc + kidx);
            }
        }
        #pragma unroll
        for (int am = 0; am < 4; ++am)
            #pragma unroll
            for (int bn = 0; bn < 2; ++bn)
                acc[am][bn] = __builtin_amdgcn_wmma_f32_16x16x32_bf16(
                    false, fa[am].v, false, fb[bn].v,
                    (short)0, acc[am][bn], false, false);
        __syncthreads();   // all waves done reading buf[cur] before it is re-filled
    }

    // C layout: VGPR r -> lanes 0-15: (M=r, N=lane); lanes 16-31: (M=8+r, N=lane-16)
    const int nloc = lane & 15;
    #pragma unroll
    for (int am = 0; am < 4; ++am) {
        int rowb = row0 + wm * 64 + am * 16 + half * 8;
        #pragma unroll
        for (int bn = 0; bn < 2; ++bn) {
            int col = col0 + wn * 32 + bn * 16 + nloc;
            float bv = bias[col];
            float* op = out + (size_t)rowb * BDIM + col;
            #pragma unroll
            for (int r = 0; r < 8; ++r)
                op[(size_t)r * BDIM] = acc[am][bn][r] + bv;
        }
    }
}

// ---------------- WKV recurrence: one thread per (batch, channel) ----------------
__global__ __launch_bounds__(256) void wkv_kernel(
    const float* __restrict__ K, const float* __restrict__ V,
    const float* __restrict__ wr, const float* __restrict__ ur,
    __bf16* __restrict__ O)
{
    int tid = blockIdx.x * blockDim.x + threadIdx.x;   // 0..8191
    int b = tid >> 10;
    int c = tid & (BDIM - 1);

    float w = wr[c];
    float u = ur[c];

    size_t base = ((size_t)b * TLEN) * BDIM + c;
    const float* kp = K + base;
    const float* vp = V + base;
    __bf16*      op = O + base;

    float pp = kp[0];
    float aa = vp[0];
    float bb = 1.0f;
    op[0] = (__bf16)vp[0];

    for (int t = 1; t < TLEN; ++t) {
        size_t off = (size_t)t * BDIM;
        float kt = kp[off];
        float vt = vp[off];
        float ww = u + kt;
        float qq = fmaxf(ww, pp);
        float e1 = __expf(pp - qq);
        float e2 = __expf(ww - qq);
        op[off] = (__bf16)((e1 * aa + e2 * vt) / (e1 * bb + e2));
        float ww2 = pp - w;
        float qq2 = fmaxf(ww2, kt);
        float e1b = __expf(ww2 - qq2);
        float e2b = __expf(kt - qq2);
        aa = e1b * aa + e2b * vt;
        bb = e1b * bb + e2b;
        pp = qq2;
    }
}

extern "C" void kernel_launch(void* const* d_in, const int* in_sizes, int n_in,
                              void* d_out, int out_size, void* d_ws, size_t ws_size,
                              hipStream_t stream) {
    (void)in_sizes; (void)n_in; (void)out_size; (void)ws_size;
    const float* x  = (const float*)d_in[0];
    // d_in[1]=Wq, d_in[2]=bq dead in reference — skipped.
    const float* Wk = (const float*)d_in[3];
    const float* bk = (const float*)d_in[4];
    const float* Wv = (const float*)d_in[5];
    const float* bv = (const float*)d_in[6];
    const float* wr = (const float*)d_in[7];
    const float* ur = (const float*)d_in[8];
    const float* Wo = (const float*)d_in[9];
    const float* bo = (const float*)d_in[10];

    char* ws = (char*)d_ws;
    const size_t MB = 1024 * 1024;
    __bf16* Xb  = (__bf16*)(ws + 0);         // 32 MB
    __bf16* Wkt = (__bf16*)(ws + 32 * MB);   // 2 MB
    __bf16* Wvt = (__bf16*)(ws + 34 * MB);   // 2 MB
    __bf16* Wot = (__bf16*)(ws + 36 * MB);   // 2 MB
    float*  Kb  = (float*) (ws + 64 * MB);   // 64 MB
    float*  Vb  = (float*) (ws + 128 * MB);  // 64 MB
    __bf16* Ob  = (__bf16*)(ws + 192 * MB);  // 32 MB

    // prep
    f32_to_bf16_kernel<<<dim3((MROWS * BDIM) / (256 * 8)), dim3(256), 0, stream>>>(x, Xb);
    dim3 tg(BDIM / 32, BDIM / 32);
    transpose_f32_bf16<<<tg, dim3(256), 0, stream>>>(Wk, Wkt);
    transpose_f32_bf16<<<tg, dim3(256), 0, stream>>>(Wv, Wvt);
    transpose_f32_bf16<<<tg, dim3(256), 0, stream>>>(Wo, Wot);

    dim3 gg(BDIM / 128, MROWS / 128);        // 8 x 128 blocks
    gemm_bf16_tn<<<gg, dim3(256), 0, stream>>>(Xb, Wkt, bk, Kb);
    gemm_bf16_tn<<<gg, dim3(256), 0, stream>>>(Xb, Wvt, bv, Vb);

    wkv_kernel<<<dim3((NBATCH * BDIM) / 256), dim3(256), 0, stream>>>(Kb, Vb, wr, ur, Ob);

    gemm_bf16_tn<<<gg, dim3(256), 0, stream>>>(Ob, Wot, bo, (float*)d_out);
}
